// BaselineGNN3D_72688026517890
// MI455X (gfx1250) — matrix operand order repaired and checked by reading, not compiled, not run on hardware
//
#include <hip/hip_runtime.h>
#include <hip/hip_bf16.h>
#include <math.h>

typedef __attribute__((ext_vector_type(16))) _Float16 v16h;
typedef __attribute__((ext_vector_type(8)))  _Float16 v8h;
typedef __attribute__((ext_vector_type(8)))  float    v8f;

#define WAVES 8
#define ACT_W 160          // max padded activation width (layer2 out: 150 -> 160)
#define NFRAG 42           // 3 + 20 + 15 + 4 WMMA B-fragments
#define WPACK_HALFS (NFRAG * 512)

__device__ __forceinline__ void wave_sync() {
    __builtin_amdgcn_wave_barrier();
    asm volatile("s_wait_dscnt 0x0" ::: "memory");
    __builtin_amdgcn_wave_barrier();
}

// Branch-free tanh: 1 v_exp_f32 + 1 v_rcp_f32 + 3 VALU, no exec divergence.
// tanh(x) = sign(x) * (1 - 2/(exp(2|x|)+1)); exp overflow -> inf -> rcp -> 0 -> 1.
__device__ __forceinline__ float tanh_fast(float x) {
    const float e = __builtin_amdgcn_exp2f(fabsf(x) * 2.88539008177793f); // 2*log2(e)
    const float r = 1.0f - 2.0f * __builtin_amdgcn_rcpf(e + 1.0f);
    return copysignf(r, x);
}

// Pack two f32 -> two f16 in a dword (RNE via scalar converts; lowers to v_pack).
__device__ __forceinline__ unsigned int pk16(float a, float b) {
    union { _Float16 h[2]; unsigned int u; } x;
    x.h[0] = (_Float16)a; x.h[1] = (_Float16)b;
    return x.u;
}

// ---------------------------------------------------------------------------
// Prep: fold BN into W1/b1, pack all weights into WMMA B-fragment layout (f16)
// B frag layout (16-bit, 32x16): lane l, packed element j (v16h index):
//   K_local = ((j>=8)?16:0) + ((l>=16)?8:0) + (j&7) ; n_local = l & 15
// Fragment order: [layer][ntile][kchunk]; frag f stores lane l's 16 halfs at
//   wpack[f*512 + l*16 + j]
// ---------------------------------------------------------------------------
__global__ void prep_kernel(const float* __restrict__ gamma, const float* __restrict__ beta,
                            const float* __restrict__ mean,  const float* __restrict__ var,
                            const float* __restrict__ W1, const float* __restrict__ b1,
                            const float* __restrict__ W2, const float* __restrict__ b2,
                            const float* __restrict__ W3, const float* __restrict__ b3,
                            const float* __restrict__ W4, const float* __restrict__ b4,
                            _Float16* __restrict__ wpack, float* __restrict__ biasws) {
    __shared__ float scale[32], shift[32];
    const int t = threadIdx.x;
    if (t < 24) {
        float s = gamma[t] * rsqrtf(var[t] + 1e-5f);
        scale[t] = s;
        shift[t] = beta[t] - mean[t] * s;
    }
    __syncthreads();

    // biases (b1 folded): [0,40) b1f | [40,190) b2 | [190,230) b3 | [230,250) b4
    if (t < 256) {
        float bv = 0.0f;
        if (t < 40) {
            float acc = b1[t];
            for (int k = 0; k < 24; ++k) acc += shift[k] * W1[k * 40 + t];
            bv = acc;
        } else if (t < 190) bv = b2[t - 40];
        else if (t < 230)   bv = b3[t - 190];
        else if (t < 250)   bv = b4[t - 230];
        biasws[t] = bv;
    }

    for (int e = t; e < WPACK_HALFS; e += blockDim.x) {
        const int f = e >> 9, r = e & 511, l = r >> 4, j = r & 15;
        const int Klocal = ((j >> 3) << 4) + ((l >> 4) << 3) + (j & 7);
        const int nlocal = l & 15;
        int nt, kc, Kact, Nact; const float* W; bool fold = false;
        if (f < 3)       { nt = f;            kc = 0;           Kact = 24;  Nact = 40;  W = W1; fold = true; }
        else if (f < 23) { int g = f - 3;  nt = g >> 1; kc = g & 1;  Kact = 40;  Nact = 150; W = W2; }
        else if (f < 38) { int g = f - 23; nt = g / 5;  kc = g % 5;  Kact = 150; Nact = 40;  W = W3; }
        else             { int g = f - 38; nt = g >> 1; kc = g & 1;  Kact = 40;  Nact = 20;  W = W4; }
        const int K = kc * 32 + Klocal, n = nt * 16 + nlocal;
        float v = 0.0f;
        if (K < Kact && n < Nact) { v = W[K * Nact + n]; if (fold) v *= scale[K]; }
        wpack[e] = (_Float16)v;
    }
}

// ---------------------------------------------------------------------------
__global__ void zero_kernel(float* __restrict__ p, int n) {
    int i = blockIdx.x * blockDim.x + threadIdx.x;
    if (i < n) p[i] = 0.0f;
}
__global__ void copy_kernel(float* __restrict__ d, const float* __restrict__ s, int n) {
    int i = blockIdx.x * blockDim.x + threadIdx.x;
    if (i < n) d[i] = s[i];
}

__global__ void efeat_kernel(const float* __restrict__ coords, const float* __restrict__ elen,
                             const float* __restrict__ evec, const int* __restrict__ nto,
                             const int* __restrict__ nfrom, float4* __restrict__ efeat, int E) {
    int e = blockIdx.x * blockDim.x + threadIdx.x;
    if (e >= E) return;
    const int f = nfrom[e], tt = nto[e];
    const float fx = coords[f * 3 + 0], fy = coords[f * 3 + 1], fz = coords[f * 3 + 2];
    const float tx = coords[tt * 3 + 0], ty = coords[tt * 3 + 1], tz = coords[tt * 3 + 2];
    const float dx = fx - tx, dy = fy - ty, dz = fz - tz;
    const float cd = fabsf(dx) + fabsf(dy) + fabsf(dz);
    const float dc = fx * tx + fy * ty + fz * tz;
    const float dd = dx * evec[e * 3 + 0] + dy * evec[e * 3 + 1] + dz * evec[e * 3 + 2];
    efeat[e] = make_float4(elen[e], cd, dc, dd);
}

// ---------------------------------------------------------------------------
// Load one A fragment k-chunk: lane's 16 K-values are two contiguous 16B runs.
//   j<8 : K = kc*32 + hi8 + j         (16B aligned)
//   j>=8: K = kc*32 + 16 + hi8 + (j-8)
// ---------------------------------------------------------------------------
__device__ __forceinline__ v16h load_afrag(const _Float16* inA, int m, int hi8, int kc) {
    const _Float16* base = inA + m * ACT_W + kc * 32 + hi8;
    const v8h lo = *(const v8h*)(base);
    const v8h hi = *(const v8h*)(base + 16);
    return __builtin_shufflevector(lo, hi, 0, 1, 2, 3, 4, 5, 6, 7,
                                   8, 9, 10, 11, 12, 13, 14, 15);
}

// ---------------------------------------------------------------------------
// One WMMA MLP layer: A (16 edges x K, f16 in LDS) x W -> tanh -> f16 LDS.
// C frag (f32 16x16): lane l, vgpr v: n = l&15 (+nt*16), m = v + ((l>>4)<<3)
// NTW >= NT: extra tiles just write zeros (pads next layer's K to 32-multiple).
// ---------------------------------------------------------------------------
template<int KCH, int NT, int NTW>
__device__ __forceinline__ void mlp_layer(const _Float16* inA, _Float16* outA,
                                          const _Float16* wlds, int fragBase,
                                          const float* bias, int Nact, int lane) {
    const int m   = lane & 15;
    const int hi8 = (lane >> 4) << 3;
    v16h a[KCH];
#pragma unroll
    for (int kc = 0; kc < KCH; ++kc) a[kc] = load_afrag(inA, m, hi8, kc);
#pragma unroll
    for (int nt = 0; nt < NTW; ++nt) {
        const int n = nt * 16 + m;
        if (nt < NT) {
            v8f c = {};
#pragma unroll
            for (int kc = 0; kc < KCH; ++kc) {
                v16h b = *(const v16h*)(wlds + (fragBase + nt * KCH + kc) * 512 + lane * 16);
                c = __builtin_amdgcn_wmma_f32_16x16x32_f16(false, a[kc], false, b,
                                                           (short)0, c, false, false);
            }
            const bool valid = (n < Nact);
            const float bn = valid ? bias[n] : 0.0f;
#pragma unroll
            for (int v = 0; v < 8; ++v) {
                float tv = valid ? tanh_fast(c[v] + bn) : 0.0f;
                outA[(v + hi8) * ACT_W + n] = (_Float16)tv;
            }
        } else {
#pragma unroll
            for (int v = 0; v < 8; ++v) outA[(v + hi8) * ACT_W + n] = (_Float16)0.0f;
        }
    }
}

// ---------------------------------------------------------------------------
// Message kernel: one wave handles 16 edges per tile, MLP via WMMA,
// atomic scatter into next-state.
// ---------------------------------------------------------------------------
__global__ void message_kernel(const float* __restrict__ Scur, float* __restrict__ Snext,
                               const float4* __restrict__ efeat,
                               const int* __restrict__ nfrom, const int* __restrict__ nto,
                               const _Float16* __restrict__ wpack, const float* __restrict__ biasws,
                               int E, int numTiles) {
    __shared__ __align__(32) _Float16 wlds[WPACK_HALFS];
    __shared__ float blds[256];
    __shared__ __align__(32) _Float16 act[WAVES][2][16 * ACT_W];
    __shared__ int tileTo[WAVES][16];

    // stage weights + biases once per block
    {
        const uint4* src = (const uint4*)wpack;
        uint4* dst = (uint4*)wlds;
        for (int i = threadIdx.x; i < WPACK_HALFS / 8; i += blockDim.x) dst[i] = src[i];
        for (int i = threadIdx.x; i < 256; i += blockDim.x) blds[i] = biasws[i];
    }
    __syncthreads();

    const int wave  = threadIdx.x >> 5;
    const int lane  = threadIdx.x & 31;
    const int gwave = blockIdx.x * WAVES + wave;
    const int nwav  = gridDim.x * WAVES;
    const int m     = lane & 15;
    const int hi8   = (lane >> 4) << 3;

    _Float16* A0 = &act[wave][0][0];
    _Float16* A1 = &act[wave][1][0];

    for (int tile = gwave; tile < numTiles; tile += nwav) {
        // ---- stage 16 edge input rows (24 values, padded to 32) ----
        // Both half-waves cooperate on row m = lane&15:
        //   half 0: state[0..11] -> cols 0..11, plus tileTo
        //   half 1: state[12..19] + edge_feat -> cols 12..23, zero cols 24..31
        {
            const int e  = tile * 16 + m;
            const bool ok = (e < E);
            _Float16* row = A0 + m * ACT_W;
            if (hi8 == 0) {
                int to = -1;
                if (ok) {
                    const int from = nfrom[e];
                    to = nto[e];
                    const float* sr = Scur + (size_t)from * 20;
                    const float4 s0 = *(const float4*)(sr + 0);
                    const float4 s1 = *(const float4*)(sr + 4);
                    const float4 s2 = *(const float4*)(sr + 8);
                    uint4 p0; p0.x = pk16(s0.x, s0.y); p0.y = pk16(s0.z, s0.w);
                              p0.z = pk16(s1.x, s1.y); p0.w = pk16(s1.z, s1.w);
                    uint2 p1; p1.x = pk16(s2.x, s2.y); p1.y = pk16(s2.z, s2.w);
                    *(uint4*)(row + 0) = p0;       // cols 0..7   (16B aligned)
                    *(uint2*)(row + 8) = p1;       // cols 8..11  (8B aligned)
                } else {
                    *(uint4*)(row + 0) = make_uint4(0, 0, 0, 0);
                    *(uint2*)(row + 8) = make_uint2(0, 0);
                }
                tileTo[wave][m] = to;
            } else {
                if (ok) {
                    const int from = nfrom[e];
                    const float* sr = Scur + (size_t)from * 20;
                    const float4 s3 = *(const float4*)(sr + 12);
                    const float4 s4 = *(const float4*)(sr + 16);
                    const float4 ef = efeat[e];
                    uint2 p2; p2.x = pk16(s3.x, s3.y); p2.y = pk16(s3.z, s3.w);
                    uint4 p3; p3.x = pk16(s4.x, s4.y); p3.y = pk16(s4.z, s4.w);
                              p3.z = pk16(ef.x, ef.y); p3.w = pk16(ef.z, ef.w);
                    *(uint2*)(row + 12) = p2;      // cols 12..15 (8B aligned)
                    *(uint4*)(row + 16) = p3;      // cols 16..23 (16B aligned)
                } else {
                    *(uint2*)(row + 12) = make_uint2(0, 0);
                    *(uint4*)(row + 16) = make_uint4(0, 0, 0, 0);
                }
                *(uint4*)(row + 24) = make_uint4(0, 0, 0, 0); // pad cols 24..31
            }
        }
        wave_sync();

        // L1: 24(pad32) -> 40   (1 kchunk, 3 tiles, pad out to 64)
        mlp_layer<1, 3, 4>(A0, A1, wlds, 0, blds + 0, 40, lane);
        wave_sync();
        // L2: 40(pad64) -> 150  (2 kchunks, 10 tiles -> 160)
        mlp_layer<2, 10, 10>(A1, A0, wlds, 3, blds + 40, 150, lane);
        wave_sync();
        // L3: 150(pad160) -> 40 (5 kchunks, 3 tiles, pad out to 64)
        mlp_layer<5, 3, 4>(A0, A1, wlds, 23, blds + 190, 40, lane);
        wave_sync();

        // L4: 40(pad64) -> 20, tanh, atomic scatter to Snext[node_to]
        {
            v16h a[2];
#pragma unroll
            for (int kc = 0; kc < 2; ++kc) a[kc] = load_afrag(A1, m, hi8, kc);
#pragma unroll
            for (int nt = 0; nt < 2; ++nt) {
                v8f c = {};
#pragma unroll
                for (int kc = 0; kc < 2; ++kc) {
                    v16h b = *(const v16h*)(wlds + (38 + nt * 2 + kc) * 512 + lane * 16);
                    c = __builtin_amdgcn_wmma_f32_16x16x32_f16(false, a[kc], false, b,
                                                               (short)0, c, false, false);
                }
                const int n = nt * 16 + m;
                if (n < 20) {
                    const float bn = blds[230 + n];
#pragma unroll
                    for (int v = 0; v < 8; ++v) {
                        const int mr = v + hi8;
                        const int to = tileTo[wave][mr];
                        if (to >= 0) {
                            const float msg = tanh_fast(c[v] + bn);
                            atomicAdd(Snext + (size_t)to * 20 + n, msg);
                        }
                    }
                }
            }
        }
        wave_sync();
    }
}

// ---------------------------------------------------------------------------
__global__ void reduce_kernel(const float* __restrict__ S, const int* __restrict__ gidx,
                              float* __restrict__ gs, int N) {
    int i = blockIdx.x * blockDim.x + threadIdx.x;
    if (i >= N * 20) return;
    const int node = i / 20, c = i % 20;
    atomicAdd(&gs[(size_t)gidx[node] * 20 + c], S[i]);
}

__global__ void out_kernel(const float* __restrict__ gs, const float* __restrict__ Wo,
                           const float* __restrict__ bo, float* __restrict__ out, int G) {
    int g = blockIdx.x * blockDim.x + threadIdx.x;
    if (g >= G) return;
    float a0 = bo[0], a1 = bo[1];
    for (int k = 0; k < 20; ++k) {
        const float s = gs[(size_t)g * 20 + k];
        a0 += s * Wo[k * 2 + 0];
        a1 += s * Wo[k * 2 + 1];
    }
    out[g * 2 + 0] = a0;
    out[g * 2 + 1] = fmaxf(a1, 0.0f) + log1pf(expf(-fabsf(a1)));  // softplus
}

// ---------------------------------------------------------------------------
extern "C" void kernel_launch(void* const* d_in, const int* in_sizes, int n_in,
                              void* d_out, int out_size, void* d_ws, size_t ws_size,
                              hipStream_t stream) {
    const float* coords = (const float*)d_in[0];
    const float* elen   = (const float*)d_in[1];
    const float* evec   = (const float*)d_in[2];
    const int*   nto    = (const int*)d_in[3];
    const int*   nfrom  = (const int*)d_in[4];
    const int*   gidx   = (const int*)d_in[5];
    const float* gamma  = (const float*)d_in[7];
    const float* beta   = (const float*)d_in[8];
    const float* mean   = (const float*)d_in[9];
    const float* var    = (const float*)d_in[10];
    const float* W1 = (const float*)d_in[11]; const float* b1 = (const float*)d_in[12];
    const float* W2 = (const float*)d_in[13]; const float* b2 = (const float*)d_in[14];
    const float* W3 = (const float*)d_in[15]; const float* b3 = (const float*)d_in[16];
    const float* W4 = (const float*)d_in[17]; const float* b4 = (const float*)d_in[18];
    const float* Wo = (const float*)d_in[19]; const float* bo = (const float*)d_in[20];

    const int N = in_sizes[0] / 3;
    const int E = in_sizes[1];
    const int G = out_size / 2;

    size_t off = 0;
    auto carve = [&](size_t bytes) -> void* {
        void* p = (char*)d_ws + off;
        off += (bytes + 255) & ~(size_t)255;
        return p;
    };
    _Float16* wpack  = (_Float16*)carve(WPACK_HALFS * sizeof(_Float16));
    float*    biasws = (float*)carve(256 * sizeof(float));
    float4*   efeat  = (float4*)carve((size_t)E * sizeof(float4));
    float*    SA     = (float*)carve((size_t)N * 20 * sizeof(float));
    float*    SB     = (float*)carve((size_t)N * 20 * sizeof(float));
    float*    GS     = (float*)carve((size_t)G * 20 * sizeof(float));

    prep_kernel<<<1, 256, 0, stream>>>(gamma, beta, mean, var, W1, b1, W2, b2, W3, b3, W4, b4,
                                       wpack, biasws);
    zero_kernel<<<(N * 20 + 255) / 256, 256, 0, stream>>>(SA, N * 20);
    efeat_kernel<<<(E + 255) / 256, 256, 0, stream>>>(coords, elen, evec, nto, nfrom, efeat, E);

    const int numTiles = (E + 15) / 16;
    float* cur = SA; float* nxt = SB;
    for (int r = 0; r < 3; ++r) {
        copy_kernel<<<(N * 20 + 255) / 256, 256, 0, stream>>>(nxt, cur, N * 20);
        message_kernel<<<1024, WAVES * 32, 0, stream>>>(cur, nxt, efeat, nfrom, nto,
                                                        wpack, biasws, E, numTiles);
        float* t = cur; cur = nxt; nxt = t;
    }

    zero_kernel<<<(G * 20 + 255) / 256, 256, 0, stream>>>(GS, G * 20);
    reduce_kernel<<<(N * 20 + 255) / 256, 256, 0, stream>>>(cur, gidx, GS, N);
    out_kernel<<<(G + 63) / 64, 64, 0, stream>>>(GS, Wo, bo, (float*)d_out, G);
}
